// MultiHeadedAttention_7962869367440
// MI455X (gfx1250) — matrix operand order repaired and checked by reading, not compiled, not run on hardware
//
#include <hip/hip_runtime.h>

// Problem constants (match reference)
#define BB   2
#define SS   2048
#define DDIM 1024
#define HH   16
#define HDIM 64

typedef _Float16 h2   __attribute__((ext_vector_type(2)));
typedef _Float16 v16h __attribute__((ext_vector_type(16)));
typedef float    v8f  __attribute__((ext_vector_type(8)));
typedef int      v4i  __attribute__((ext_vector_type(4)));

#if __has_builtin(__builtin_amdgcn_global_load_async_to_lds_b128)
#define HAVE_ASYNC 1
#else
#define HAVE_ASYNC 0
#endif

// 16B global -> LDS tile copy. Async path bypasses VGPRs (ASYNCcnt-tracked,
// enables double buffering); fallback is a plain load+store.
// Builtin signature (from hipcc diagnostic): (v4i AS1* gsrc, v4i AS3* ldsdst,
// imm offset, imm cpol).
__device__ __forceinline__ void ld_tile16(void* lds, const void* g) {
#if HAVE_ASYNC
  typedef __attribute__((address_space(1))) v4i* gptr_t;
  typedef __attribute__((address_space(3))) v4i* lptr_t;
  __builtin_amdgcn_global_load_async_to_lds_b128((gptr_t)(void*)g,
                                                 (lptr_t)lds, 0, 0);
#else
  *(uint4*)lds = *(const uint4*)g;
#endif
}

__device__ __forceinline__ void wait_async() {
#if HAVE_ASYNC
#if __has_builtin(__builtin_amdgcn_s_wait_asynccnt)
  __builtin_amdgcn_s_wait_asynccnt(0);
#else
  asm volatile("s_wait_asynccnt 0" ::: "memory");
#endif
#endif
}

__device__ __forceinline__ v8f zero8() {
  v8f z = {0.f,0.f,0.f,0.f,0.f,0.f,0.f,0.f};
  return z;
}

__device__ __forceinline__ v8f wmma32(v16h a, v16h b, v8f c) {
  // D = A(16x32 f16) * B(32x16 f16) + C(16x16 f32)
  return __builtin_amdgcn_wmma_f32_16x16x32_f16(false, a, false, b, (short)0, c,
                                                false, false);
}

// ---- WMMA fragment loaders (contiguous k order: k = 16*(lane/16) + s, same in A and B) ----

// A fragment: element (m,k) at src[m*lda + k]; lda even, src 4B aligned.
__device__ __forceinline__ v16h load_frag_a(const _Float16* src, int lda) {
  const int lane = threadIdx.x & 31;
  const int m    = lane & 15;
  const int hi   = (lane >> 4) & 1;
  const _Float16* row = src + m * lda;
  v16h a;
#pragma unroll
  for (int i = 0; i < 4; ++i) {
    const int k0 = 2 * i + hi * 16;
    h2 p = *(const h2*)(row + k0);
    h2 q = *(const h2*)(row + 8 + k0);
    a[2 * i]     = p[0];
    a[2 * i + 1] = p[1];
    a[8 + 2 * i]     = q[0];
    a[8 + 2 * i + 1] = q[1];
  }
  return a;
}

// B fragment: element (k,n) at src[k*ldb + n] (row-major KxN)
__device__ __forceinline__ v16h load_frag_b(const _Float16* src, int ldb) {
  const int lane = threadIdx.x & 31;
  const int n    = lane & 15;
  const int hi   = (lane >> 4) & 1;
  v16h b;
#pragma unroll
  for (int i = 0; i < 8; ++i) {
    const int k = 2 * i + hi * 16;
    b[2 * i]     = src[k * ldb + n];
    b[2 * i + 1] = src[(k + 1) * ldb + n];
  }
  return b;
}

// B fragment from transposed storage: element (k,n) at src[n*ld + k]
__device__ __forceinline__ v16h load_frag_bt(const _Float16* src, int ld) {
  const int lane = threadIdx.x & 31;
  const int n    = lane & 15;
  const int hi   = (lane >> 4) & 1;
  const _Float16* row = src + n * ld;
  v16h b;
#pragma unroll
  for (int i = 0; i < 8; ++i) {
    const int k0 = 2 * i + hi * 16;
    h2 p = *(const h2*)(row + k0);
    b[2 * i]     = p[0];
    b[2 * i + 1] = p[1];
  }
  return b;
}

// 16-lane-half reductions (wave32: rows r / r+8 live in the two lane halves)
__device__ __forceinline__ float hmax16(float v) {
#pragma unroll
  for (int off = 1; off < 16; off <<= 1) v = fmaxf(v, __shfl_xor(v, off, 32));
  return v;
}
__device__ __forceinline__ float hsum16(float v) {
#pragma unroll
  for (int off = 1; off < 16; off <<= 1) v += __shfl_xor(v, off, 32);
  return v;
}

// ---------------- Stage 0: f32 -> f16 convert ----------------
__global__ __launch_bounds__(256) void k_cvt(const float* __restrict__ in,
                                             _Float16* __restrict__ out, int n4) {
  int i = blockIdx.x * blockDim.x + threadIdx.x;
  if (i >= n4) return;
  float4 v = ((const float4*)in)[i];
  h2 a = {(_Float16)v.x, (_Float16)v.y};
  h2 b = {(_Float16)v.z, (_Float16)v.w};
  ((h2*)out)[2 * i]     = a;
  ((h2*)out)[2 * i + 1] = b;
}

// ---------------- Stage 0b: weight transpose [H,D,HD] -> [D, H*HD] f16 ----
__global__ __launch_bounds__(256) void k_wt(const float* __restrict__ in,
                                            _Float16* __restrict__ out) {
  int idx = blockIdx.x * blockDim.x + threadIdx.x;  // H*D*HD/2 threads
  int k = (idx & (HDIM / 2 - 1)) * 2;
  int d = (idx >> 5) & (DDIM - 1);
  int h = idx >> 15;
  float2 v = *(const float2*)(in + (((size_t)h * DDIM + d) * HDIM + k));
  _Float16* q = out + ((size_t)d * DDIM + h * HDIM + k);
  q[0] = (_Float16)v.x;
  q[1] = (_Float16)v.y;
}

// ---------------- GEMM: C[M=4096,N=1024] = A[M,1024] * B[1024,N] + bias ----
// MODE 0: out f16 scattered into [B,H,S,HD]   (projection)
// MODE 1: out f32 row-major [M,N]             (output projection -> d_out)
template <int MODE>
__global__ __launch_bounds__(256) void gemm_kernel(
    const _Float16* __restrict__ A, const _Float16* __restrict__ Bm,
    const float* __restrict__ bias, void* __restrict__ out) {
  constexpr int K = DDIM, N = DDIM;
  constexpr int LDA = 40;   // 32 + 8 pad: 80B rows (16B multiple, conflict-free)
  constexpr int LDB = 136;  // 128 + 8 pad: 272B rows
  __shared__ __align__(16) _Float16 As[2][128 * LDA];
  __shared__ __align__(16) _Float16 Bs[2][32 * LDB];

  const int tid  = threadIdx.x;
  const int wave = tid >> 5;
  const int bm = blockIdx.y * 128;
  const int bn = blockIdx.x * 128;
  const int wm = (wave >> 2) * 64;  // 2 waves in M
  const int wn = (wave & 3) * 32;   // 4 waves in N

  v8f acc[4][2];
#pragma unroll
  for (int i = 0; i < 4; ++i) {
    acc[i][0] = zero8();
    acc[i][1] = zero8();
  }

  auto stage = [&](int kk, int buf) {
#pragma unroll
    for (int c = 0; c < 2; ++c) {
      int id = tid + c * 256;
      int ra = id >> 2, ca = (id & 3) * 8;
      ld_tile16(&As[buf][ra * LDA + ca],
                &A[(size_t)(bm + ra) * K + kk + ca]);
      int rb = id >> 4, cb = (id & 15) * 8;
      ld_tile16(&Bs[buf][rb * LDB + cb],
                &Bm[(size_t)(kk + rb) * N + bn + cb]);
    }
  };

  stage(0, 0);
  wait_async();
  __syncthreads();

  int cur = 0;
  for (int kk = 0; kk < K; kk += 32) {
    if (kk + 32 < K) stage(kk + 32, cur ^ 1);  // overlap next tile with WMMA

    v16h bfr0 = load_frag_b(&Bs[cur][wn], LDB);
    v16h bfr1 = load_frag_b(&Bs[cur][wn + 16], LDB);
#pragma unroll
    for (int i = 0; i < 4; ++i) {
      v16h afr = load_frag_a(&As[cur][(wm + i * 16) * LDA], LDA);
      acc[i][0] = wmma32(afr, bfr0, acc[i][0]);
      acc[i][1] = wmma32(afr, bfr1, acc[i][1]);
    }
    wait_async();
    __syncthreads();
    cur ^= 1;
  }

  // Epilogue
  const int lane = tid & 31;
  const int n0   = lane & 15;
  const int hi   = (lane >> 4) & 1;
#pragma unroll
  for (int i = 0; i < 4; ++i) {
#pragma unroll
    for (int j = 0; j < 2; ++j) {
#pragma unroll
      for (int r = 0; r < 8; ++r) {
        int m = bm + wm + i * 16 + r + hi * 8;
        int n = bn + wn + j * 16 + n0;
        float v = acc[i][j][r] + bias[n];
        if (MODE == 0) {
          // [B,S] x [H*HD]  ->  [B,H,S,HD]
          int b = m >> 11, s = m & (SS - 1), h = n >> 6, k2 = n & (HDIM - 1);
          ((_Float16*)out)[(((size_t)(b * HH + h) * SS + s) * HDIM) + k2] =
              (_Float16)v;
        } else {
          ((float*)out)[(size_t)m * N + n] = v;
        }
      }
    }
  }
}

// ---------------- Flash attention over f16 [B,H,S,HD] ----------------
__global__ __launch_bounds__(256) void attn_kernel(
    const _Float16* __restrict__ Qp, const _Float16* __restrict__ Kp,
    const _Float16* __restrict__ Vp, _Float16* __restrict__ ctx) {
  constexpr int HDP = 72;  // 144B rows: 16B multiple, conflict-free bt reads
  constexpr int PLD = 40;  // P-buffer row stride
  __shared__ __align__(16) _Float16 Ks[2][64 * HDP];
  __shared__ __align__(16) _Float16 Vs[2][64 * HDP];
  __shared__ __align__(16) _Float16 Ps[8 * 16 * PLD];

  const int tid  = threadIdx.x;
  const int wave = tid >> 5;
  const int lane = tid & 31;
  const int n0   = lane & 15;
  const int hi   = (lane >> 4) & 1;
  const int bh   = blockIdx.y;                 // b*H + h
  const size_t base = (size_t)bh * SS * HDIM;
  const int qrow0 = blockIdx.x * 128 + wave * 16;

  // Preload this wave's Q fragments; fold in softmax scale 1/sqrt(64)=0.125
  v16h qf0 = load_frag_a(Qp + base + (size_t)qrow0 * HDIM, HDIM);
  v16h qf1 = load_frag_a(Qp + base + (size_t)qrow0 * HDIM + 32, HDIM);
  qf0 = qf0 * (_Float16)0.125f;
  qf1 = qf1 * (_Float16)0.125f;

  v8f acc[4];
#pragma unroll
  for (int i = 0; i < 4; ++i) acc[i] = zero8();
  float mrow[8], lrow[8];
#pragma unroll
  for (int r = 0; r < 8; ++r) {
    mrow[r] = -1e30f;
    lrow[r] = 0.f;
  }
  _Float16* Pw = &Ps[wave * 16 * PLD];

  auto stage_kv = [&](int kt, int buf) {
#pragma unroll
    for (int c = 0; c < 2; ++c) {
      int id = tid + c * 256;
      int row = id >> 3;
      int col = (id & 7) * 8;
      ld_tile16(&Ks[buf][row * HDP + col],
                &Kp[base + (size_t)(kt + row) * HDIM + col]);
      ld_tile16(&Vs[buf][row * HDP + col],
                &Vp[base + (size_t)(kt + row) * HDIM + col]);
    }
  };

  stage_kv(0, 0);
  wait_async();
  __syncthreads();

  int cur = 0;
  for (int kt = 0; kt < SS; kt += 64) {
    if (kt + 64 < SS) stage_kv(kt + 64, cur ^ 1);  // overlap with compute

    // Scores S[16 x 64] = Q(16x64) * Ktile^T: 4 n-tiles x 2 k-steps
    v8f sf[4];
#pragma unroll
    for (int j = 0; j < 4; ++j) {
      v8f s = zero8();
      s = wmma32(qf0, load_frag_bt(&Ks[cur][(j * 16) * HDP], HDP), s);
      s = wmma32(qf1, load_frag_bt(&Ks[cur][(j * 16) * HDP + 32], HDP), s);
      sf[j] = s;
    }

    // Online softmax (fragment row r -> matrix rows r / r+8 by lane half)
#pragma unroll
    for (int r = 0; r < 8; ++r) {
      float mx = sf[0][r];
      mx = fmaxf(mx, sf[1][r]);
      mx = fmaxf(mx, sf[2][r]);
      mx = fmaxf(mx, sf[3][r]);
      mx = hmax16(mx);
      float mnew  = fmaxf(mrow[r], mx);
      float alpha = __expf(mrow[r] - mnew);
      mrow[r] = mnew;
      float rsum = 0.f;
#pragma unroll
      for (int j = 0; j < 4; ++j) {
        float p  = __expf(sf[j][r] - mnew);
        sf[j][r] = p;
        rsum += p;
      }
      rsum = hsum16(rsum);
      lrow[r] = lrow[r] * alpha + rsum;
#pragma unroll
      for (int jj = 0; jj < 4; ++jj) acc[jj][r] = acc[jj][r] * alpha;
      // Spill probabilities to per-wave LDS buffer (A-fragment source)
#pragma unroll
      for (int j = 0; j < 4; ++j)
        Pw[(r + hi * 8) * PLD + j * 16 + n0] = (_Float16)sf[j][r];
    }
    // Same-wave DS ops are in-order; just drain the DS counter.
    asm volatile("s_wait_dscnt 0" ::: "memory");

    // acc(16x64) += P(16x64) * Vtile(64x64)
    v16h pa0 = load_frag_a(Pw, PLD);
    v16h pa1 = load_frag_a(Pw + 32, PLD);
#pragma unroll
    for (int jj = 0; jj < 4; ++jj) {
      acc[jj] = wmma32(pa0, load_frag_b(&Vs[cur][jj * 16], HDP), acc[jj]);
      acc[jj] =
          wmma32(pa1, load_frag_b(&Vs[cur][32 * HDP + jj * 16], HDP), acc[jj]);
    }
    wait_async();
    __syncthreads();
    cur ^= 1;
  }

  // Epilogue: normalize, write ctx as f16 [B, S, H*HD]
  const int b = bh >> 4, h = bh & 15;
#pragma unroll
  for (int jj = 0; jj < 4; ++jj) {
#pragma unroll
    for (int r = 0; r < 8; ++r) {
      int srow = qrow0 + r + hi * 8;
      float v  = acc[jj][r] / lrow[r];
      ctx[((size_t)b * SS + srow) * DDIM + h * HDIM + jj * 16 + n0] =
          (_Float16)v;
    }
  }
}

// ---------------- Host side ----------------
extern "C" void kernel_launch(void* const* d_in, const int* in_sizes, int n_in,
                              void* d_out, int out_size, void* d_ws,
                              size_t ws_size, hipStream_t stream) {
  const float* query = (const float*)d_in[0];
  const float* key   = (const float*)d_in[1];
  const float* value = (const float*)d_in[2];
  const float* Wq = (const float*)d_in[3];
  const float* bq = (const float*)d_in[4];
  const float* Wk = (const float*)d_in[5];
  const float* bk = (const float*)d_in[6];
  const float* Wv = (const float*)d_in[7];
  const float* bv = (const float*)d_in[8];
  const float* Wo = (const float*)d_in[9];
  const float* bo = (const float*)d_in[10];
  float* out = (float*)d_out;

  const size_t XEL = (size_t)BB * SS * DDIM;  // 4,194,304
  const size_t WEL = (size_t)DDIM * DDIM;     // 1,048,576

  _Float16* ws   = (_Float16*)d_ws;           // 64 MB total
  _Float16* q16  = ws;
  _Float16* k16  = q16 + XEL;
  _Float16* v16  = k16 + XEL;
  _Float16* wq16 = v16 + XEL;
  _Float16* wk16 = wq16 + WEL;
  _Float16* wv16 = wk16 + WEL;
  _Float16* wo16 = wv16 + WEL;
  _Float16* Qp   = wo16 + WEL;  // [B,H,S,HD]
  _Float16* Kp   = Qp + XEL;
  _Float16* Vp   = Kp + XEL;
  _Float16* c16  = Vp + XEL;    // ctx [B,S,D]

  // Stage 0: converts
  {
    int n4 = (int)(XEL / 4);
    k_cvt<<<n4 / 256, 256, 0, stream>>>(query, q16, n4);
    k_cvt<<<n4 / 256, 256, 0, stream>>>(key, k16, n4);
    k_cvt<<<n4 / 256, 256, 0, stream>>>(value, v16, n4);
    int w4 = (int)(WEL / 4);
    k_cvt<<<w4 / 256, 256, 0, stream>>>(Wo, wo16, w4);
    int wt = HH * DDIM * HDIM / 2;  // 524,288 threads
    k_wt<<<wt / 256, 256, 0, stream>>>(Wq, wq16);
    k_wt<<<wt / 256, 256, 0, stream>>>(Wk, wk16);
    k_wt<<<wt / 256, 256, 0, stream>>>(Wv, wv16);
  }

  // Stage 1: Q/K/V projections -> [B,H,S,HD] f16
  dim3 gg(DDIM / 128, (BB * SS) / 128);  // (8, 32)
  gemm_kernel<0><<<gg, 256, 0, stream>>>(q16, wq16, bq, (void*)Qp);
  gemm_kernel<0><<<gg, 256, 0, stream>>>(k16, wk16, bk, (void*)Kp);
  gemm_kernel<0><<<gg, 256, 0, stream>>>(v16, wv16, bv, (void*)Vp);

  // Stage 2: fused flash attention -> ctx f16 [B,S,D]
  dim3 ga(SS / 128, BB * HH);  // (16, 32)
  attn_kernel<<<ga, 256, 0, stream>>>(Qp, Kp, Vp, c16);

  // Stage 3: output projection -> f32 d_out
  gemm_kernel<1><<<gg, 256, 0, stream>>>(c16, wo16, bo, (void*)out);
}
